// DecoderRNN_53644141527507
// MI455X (gfx1250) — matrix operand order repaired and checked by reading, not compiled, not run on hardware
//
#include <hip/hip_runtime.h>
#include <hip/hip_bf16.h>

// ---------------------------------------------------------------------------
// DecoderRNN (teacher-forced GRU decoder) for gfx1250 / MI455X.
// Strategy:
//   * precompute step-invariant z-contributions (GIZ, OZ) and relu'd bf16
//     embedding table, bf16 weight copies (L2-resident, ~15 MB)
//   * per step: one fused WMMA kernel computes all 6 gate GEMM accumulators
//     (i_r,i_z,i_n from gathered embeddings; h_r,h_z,h_n from h_{t-1}) with
//     v_wmma_f32_16x16x32_bf16 and applies the GRU nonlinearity in-register
//   * per step: one WMMA kernel for o_t = h_t . Wout_hᵀ + OZ -> d_out
// ---------------------------------------------------------------------------

typedef __bf16 bf16;
typedef __attribute__((ext_vector_type(16))) __bf16 v16bf;
typedef __attribute__((ext_vector_type(8)))  __bf16 v8bf;
typedef __attribute__((ext_vector_type(8)))  float  v8f;

constexpr int B_  = 256;
constexpr int NS_ = 128;
constexpr int I_  = 128;
constexpr int H_  = 1024;
constexpr int V_  = 1024;
constexpr int H3_ = 3 * H_;
constexpr int HI_ = H_ + I_;   // 1152

// ---------------- WMMA helpers --------------------------------------------

__device__ __forceinline__ v8f wmma_bf16(v16bf a, v16bf b, v8f c) {
    // D = A(16x32 bf16) x B(32x16 bf16) + C(16x16 f32)
    return __builtin_amdgcn_wmma_f32_16x16x32_bf16(
        /*neg_a=*/false, a, /*neg_b=*/false, b,
        /*c_mod=*/(short)0, c, /*reuse_a=*/false, /*reuse_b=*/false);
}

// A-matrix 16x32 bf16 per-lane fragment: lane m = L&15,
// contiguous k-run {0..15} for lanes 0-15, {16..31} for lanes 16-31.
__device__ __forceinline__ v16bf make_a(const bf16* __restrict__ arow,
                                        int k0, int khalf) {
    v16bf r = *(const v16bf*)(arow + k0 + 2 * khalf);
    return r;
}

// B-matrix 32x16 bf16 per-lane fragment from row-major W[N,K]:
// lane n = L&15, contiguous K-run of 16 starting at (L>>4)*16.
// wb must already point at W + (n0 + (lane&15))*ld + ((lane>>4)<<4).
__device__ __forceinline__ v16bf make_b(const bf16* __restrict__ wb, int k0) {
    return *(const v16bf*)(wb + k0);
}

__device__ __forceinline__ float sigmoidf_(float x) {
    return 1.0f / (1.0f + __expf(-x));
}

// ---------------- fused GRU gate kernel (per step) ------------------------
// grid: (H/128, B/16), block 256 threads = 8 waves, 1 wave per 16x16 h-tile.

__global__ __launch_bounds__(256)
void gru_gate_kernel(const bf16*  __restrict__ embT,   // [V,H] relu(embed) bf16
                     const int*   __restrict__ inputs, // [NSTEPS,B]
                     int t,
                     const bf16*  __restrict__ wih,    // [3H, HI] bf16
                     const bf16*  __restrict__ whh,    // [3H, H]  bf16
                     const float* __restrict__ giz,    // [B, 3H] z-part + b_ih
                     const float* __restrict__ bhh,    // [3H]
                     const float* __restrict__ hprev,  // [B,H] f32
                     const bf16*  __restrict__ hprevb, // [B,H] bf16
                     float*       __restrict__ hnew,   // [B,H] f32
                     bf16*        __restrict__ hnewb)  // [B,H] bf16
{
    const int lane  = threadIdx.x & 31;
    const int wave  = threadIdx.x >> 5;
    const int m0    = blockIdx.y * 16;
    const int n0    = (blockIdx.x * 8 + wave) * 16;
    const int khalf = (lane >> 4) << 3;   // A fragment k-half
    const int nl    = lane & 15;          // B fragment column
    const int kb    = (lane >> 4) << 4;   // B fragment k-run base

    // Gathered embedding row for this lane's A row (teacher forcing).
    const int b0  = m0 + (lane & 15);
    const int tok = (t == 0) ? 0 : inputs[(t - 1) * B_ + b0];
    const bf16* arow_e = embT   + (size_t)tok * H_;
    const bf16* arow_h = hprevb + (size_t)b0  * H_;

    // B-operand bases: gate g row-band g*H of each weight matrix.
    const bf16* wb_r = wih + (size_t)(0 * H_ + n0 + nl) * HI_ + kb;
    const bf16* wb_z = wih + (size_t)(1 * H_ + n0 + nl) * HI_ + kb;
    const bf16* wb_n = wih + (size_t)(2 * H_ + n0 + nl) * HI_ + kb;
    const bf16* ub_r = whh + (size_t)(0 * H_ + n0 + nl) * H_  + kb;
    const bf16* ub_z = whh + (size_t)(1 * H_ + n0 + nl) * H_  + kb;
    const bf16* ub_n = whh + (size_t)(2 * H_ + n0 + nl) * H_  + kb;

    v8f air = {}, aiz = {}, ain = {}, ahr = {}, ahz = {}, ahn = {};

#pragma unroll 2
    for (int k0 = 0; k0 < H_; k0 += 32) {
        v16bf ae = make_a(arow_e, k0, khalf);
        v16bf ah = make_a(arow_h, k0, khalf);
        air = wmma_bf16(ae, make_b(wb_r, k0), air);
        aiz = wmma_bf16(ae, make_b(wb_z, k0), aiz);
        ain = wmma_bf16(ae, make_b(wb_n, k0), ain);
        ahr = wmma_bf16(ah, make_b(ub_r, k0), ahr);
        ahz = wmma_bf16(ah, make_b(ub_z, k0), ahz);
        ahn = wmma_bf16(ah, make_b(ub_n, k0), ahn);
    }

    // C/D layout: lanes 0-15 -> N=lane, M=j; lanes 16-31 -> N=lane-16, M=8+j.
    const int cn = n0 + (lane & 15);
    const int mo = (lane >> 4) * 8;
    const float bh_r = bhh[cn];
    const float bh_z = bhh[H_ + cn];
    const float bh_n = bhh[2 * H_ + cn];

#pragma unroll
    for (int j = 0; j < 8; ++j) {
        const int bb = m0 + mo + j;
        const size_t gz = (size_t)bb * H3_ + cn;
        float ir  = air[j] + giz[gz];
        float iz  = aiz[j] + giz[gz + H_];
        float inn = ain[j] + giz[gz + 2 * H_];
        float hr  = ahr[j] + bh_r;
        float hz  = ahz[j] + bh_z;
        float hn  = ahn[j] + bh_n;
        float r   = sigmoidf_(ir + hr);
        float zg  = sigmoidf_(iz + hz);
        float nn  = tanhf(inn + r * hn);
        float hp  = hprev[(size_t)bb * H_ + cn];
        float hv  = (1.0f - zg) * nn + zg * hp;
        hnew [(size_t)bb * H_ + cn] = hv;
        hnewb[(size_t)bb * H_ + cn] = (bf16)hv;
    }
}

// ---------------- output GEMM kernel (per step) ---------------------------
// o_t[b,v] = h_t . Wout_hᵀ + OZ[b,v];  grid (V/128, B/16), 8 waves/block.

__global__ __launch_bounds__(256)
void out_gemm_kernel(const bf16*  __restrict__ hb,  // [B,H] bf16
                     const bf16*  __restrict__ wo,  // [V,H] bf16 (h-columns)
                     const float* __restrict__ oz,  // [B,V] z-part + out_b
                     float*       __restrict__ out, // [B,NSTEPS,V]
                     int t)
{
    const int lane  = threadIdx.x & 31;
    const int wave  = threadIdx.x >> 5;
    const int m0    = blockIdx.y * 16;
    const int n0    = (blockIdx.x * 8 + wave) * 16;
    const int khalf = (lane >> 4) << 3;
    const int nl    = lane & 15;
    const int kb    = (lane >> 4) << 4;

    const bf16* arow = hb + (size_t)(m0 + (lane & 15)) * H_;
    const bf16* wb   = wo + (size_t)(n0 + nl) * H_ + kb;

    v8f acc = {};
#pragma unroll 4
    for (int k0 = 0; k0 < H_; k0 += 32) {
        acc = wmma_bf16(make_a(arow, k0, khalf), make_b(wb, k0), acc);
    }

    const int cn = n0 + (lane & 15);
    const int mo = (lane >> 4) * 8;
#pragma unroll
    for (int j = 0; j < 8; ++j) {
        const int bb = m0 + mo + j;
        out[(size_t)bb * NS_ * V_ + (size_t)t * V_ + cn] =
            acc[j] + oz[(size_t)bb * V_ + cn];
    }
}

// ---------------- one-time prep kernels -----------------------------------

// f32 -> bf16 (optionally relu), with independent source leading dimension
// (used to slice out_w[:, :H]).
__global__ void cvt_bf16_kernel(const float* __restrict__ src,
                                bf16* __restrict__ dst,
                                int rows, int src_ld, int cols, int relu)
{
    const size_t total = (size_t)rows * cols;
    for (size_t i = (size_t)blockIdx.x * blockDim.x + threadIdx.x;
         i < total; i += (size_t)gridDim.x * blockDim.x) {
        const int r = (int)(i / cols);
        const int c = (int)(i % cols);
        float v = src[(size_t)r * src_ld + c];
        if (relu) v = fmaxf(v, 0.0f);
        dst[i] = (bf16)v;
    }
}

// dst[b,n] = z[b,:] . W[n, coff:coff+I] + bias[n]   (K = 128, one-time, f32)
__global__ void zgemm_kernel(const float* __restrict__ z,    // [B, I]
                             const float* __restrict__ w,
                             const float* __restrict__ bias,
                             float* __restrict__ dst,        // [B, N]
                             bf16*  __restrict__ dstb,       // optional bf16 copy
                             int N, int w_ld, int coff)
{
    const int idx = blockIdx.x * blockDim.x + threadIdx.x;
    if (idx >= B_ * N) return;
    const int b = idx / N;
    const int n = idx % N;
    const float* zr = z + (size_t)b * I_;
    const float* wr = w + (size_t)n * w_ld + coff;
    float acc = bias[n];
#pragma unroll 4
    for (int k = 0; k < I_; k += 4) {
        acc += zr[k]     * wr[k]     + zr[k + 1] * wr[k + 1]
             + zr[k + 2] * wr[k + 2] + zr[k + 3] * wr[k + 3];
    }
    dst[idx] = acc;
    if (dstb) dstb[idx] = (bf16)acc;
}

// ---------------- host launch ---------------------------------------------

extern "C" void kernel_launch(void* const* d_in, const int* in_sizes, int n_in,
                              void* d_out, int out_size, void* d_ws, size_t ws_size,
                              hipStream_t stream)
{
    (void)in_sizes; (void)n_in; (void)out_size; (void)ws_size;

    const float* z       = (const float*)d_in[0];
    const int*   inputs  = (const int*)  d_in[1];
    // d_in[2] = temperature (teacher forcing always active -> unused)
    const float* embed_w = (const float*)d_in[3];
    const float* z2h_w   = (const float*)d_in[4];
    const float* z2h_b   = (const float*)d_in[5];
    const float* w_ih    = (const float*)d_in[6];
    const float* b_ih    = (const float*)d_in[7];
    const float* w_hh    = (const float*)d_in[8];
    const float* b_hh    = (const float*)d_in[9];
    const float* out_w   = (const float*)d_in[10];
    const float* out_b   = (const float*)d_in[11];
    float* out = (float*)d_out;

    // carve workspace (256B aligned slabs, ~25 MB total)
    char* p = (char*)d_ws;
    auto carve = [&](size_t bytes) {
        char* r = p;
        p += (bytes + 255) & ~(size_t)255;
        return r;
    };
    bf16*  wih_b = (bf16*)carve((size_t)H3_ * HI_ * sizeof(bf16));
    bf16*  whh_b = (bf16*)carve((size_t)H3_ * H_  * sizeof(bf16));
    bf16*  woh_b = (bf16*)carve((size_t)V_  * H_  * sizeof(bf16));
    bf16*  emb_b = (bf16*)carve((size_t)V_  * H_  * sizeof(bf16));
    float* giz   = (float*)carve((size_t)B_ * H3_ * sizeof(float));
    float* oz    = (float*)carve((size_t)B_ * V_  * sizeof(float));
    float* hf0   = (float*)carve((size_t)B_ * H_  * sizeof(float));
    float* hf1   = (float*)carve((size_t)B_ * H_  * sizeof(float));
    bf16*  hb0   = (bf16*) carve((size_t)B_ * H_  * sizeof(bf16));
    bf16*  hb1   = (bf16*) carve((size_t)B_ * H_  * sizeof(bf16));
    float* hf[2] = { hf0, hf1 };
    bf16*  hb[2] = { hb0, hb1 };

    // --- one-time: bf16 weight copies (+ relu'd embedding table) ---
    cvt_bf16_kernel<<<2048, 256, 0, stream>>>(w_ih,  wih_b, H3_, HI_, HI_, 0);
    cvt_bf16_kernel<<<2048, 256, 0, stream>>>(w_hh,  whh_b, H3_, H_,  H_,  0);
    cvt_bf16_kernel<<<1024, 256, 0, stream>>>(out_w, woh_b, V_,  HI_, H_,  0); // cols [0,H)
    cvt_bf16_kernel<<<1024, 256, 0, stream>>>(embed_w, emb_b, V_, H_, H_, 1);  // relu

    // --- one-time: step-invariant z contributions + h0 ---
    // GIZ = z . w_ih[:, H:]ᵀ + b_ih
    zgemm_kernel<<<(B_ * H3_ + 255) / 256, 256, 0, stream>>>(
        z, w_ih, b_ih, giz, nullptr, H3_, HI_, H_);
    // OZ = z . out_w[:, H:]ᵀ + out_b
    zgemm_kernel<<<(B_ * V_ + 255) / 256, 256, 0, stream>>>(
        z, out_w, out_b, oz, nullptr, V_, HI_, H_);
    // h0 = z . z2h_wᵀ + z2h_b  (f32 + bf16 copies)
    zgemm_kernel<<<(B_ * H_ + 255) / 256, 256, 0, stream>>>(
        z, z2h_w, z2h_b, hf[0], hb[0], H_, I_, 0);

    // --- sequential decode: 128 steps, 2 WMMA kernels per step ---
    const dim3 grid_gate(H_ / 128, B_ / 16);  // (8,16) blocks, 8 waves each
    const dim3 grid_out (V_ / 128, B_ / 16);
    for (int t = 0; t < NS_; ++t) {
        const float* hp  = hf[t & 1];
        const bf16*  hpb = hb[t & 1];
        float*       hn  = hf[(t + 1) & 1];
        bf16*        hnb = hb[(t + 1) & 1];
        gru_gate_kernel<<<grid_gate, 256, 0, stream>>>(
            emb_b, inputs, t, wih_b, whh_b, giz, b_hh, hp, hpb, hn, hnb);
        out_gemm_kernel<<<grid_out, 256, 0, stream>>>(
            hnb, woh_b, oz, out, t);
    }
}
